// MHAttention_18528488914991
// MI455X (gfx1250) — compile-verified, matrix operand order checked by reading, not run on hardware
//
#include <hip/hip_runtime.h>
#include <hip/hip_bf16.h>
#include <cstdint>

// ---------------------------------------------------------------------------
// MHA forward for MI455X (gfx1250, wave32, WMMA + async-to-LDS).
//   B=2, S=2048, H=1024, NH=16, DK=64
// Pipeline:
//   1) f32->f16 converts (q,k,v, Wq,Wk,Wv,Wo)
//   2) WMMA GEMM (double-buffered LDS, GLOBAL_LOAD_ASYNC_TO_LDS_B128):
//      Qh/Kh -> [b,h,s,d] f16 ; Vh -> transposed [b,h,d,s] f16
//   3) attention: scores via v_wmma_f32_16x16x32_f16, softmax in LDS,
//      attn written once to d_out, PV via WMMA -> ctx f16
//   4) WMMA GEMM: out = ctx@Wo^T+b -> f32 d_out
// ---------------------------------------------------------------------------

typedef __attribute__((ext_vector_type(16))) _Float16 v16h;
typedef __attribute__((ext_vector_type(8)))  _Float16 v8h;
typedef __attribute__((ext_vector_type(8)))  float    v8f;

#define BATCH 2
#define SEQ   2048
#define HID   1024
#define NHEAD 16
#define DK    64

__device__ __forceinline__ v16h frag_from_halves(const _Float16* p) {
  // p points at this lane's first k element (hs*8 already applied).
  // halves [0..7] from p, [8..15] from p+16 (ISA 16-bit A/B fragment layout).
  v8h lo = *(const v8h*)(p);
  v8h hi = *(const v8h*)(p + 16);
  v16h r;
#pragma unroll
  for (int i = 0; i < 8; ++i) { r[i] = lo[i]; r[8 + i] = hi[i]; }
  return r;
}

__device__ __forceinline__ v8f wmma_f16(v16h a, v16h b, v8f c) {
  return __builtin_amdgcn_wmma_f32_16x16x32_f16(false, a, false, b,
                                                (short)0, c, false, false);
}

// ---- CDNA5 async global->LDS copy (tracked by ASYNCcnt) -------------------
__device__ __forceinline__ void async_ld_b128(void* lds_ptr, const void* gptr) {
  // LDS aperture maps addr[31:0] -> LDS byte offset.
  uint32_t lds_off = (uint32_t)(uintptr_t)lds_ptr;
  asm volatile("global_load_async_to_lds_b128 %0, %1, off"
               :
               : "v"(lds_off), "v"(gptr)
               : "memory");
}
__device__ __forceinline__ void wait_async_le4() {
  asm volatile("s_wait_asynccnt 0x4" ::: "memory");
}
__device__ __forceinline__ void wait_async_0() {
  asm volatile("s_wait_asynccnt 0x0" ::: "memory");
}

// ---------------------------------------------------------------------------
// f32 -> f16 convert
// ---------------------------------------------------------------------------
__global__ void cvt_f32_to_f16(const float* __restrict__ src,
                               _Float16* __restrict__ dst, int n) {
  int i = blockIdx.x * blockDim.x + threadIdx.x;
  int stride = gridDim.x * blockDim.x;
  for (; i < n; i += stride) dst[i] = (_Float16)src[i];
}

// ---------------------------------------------------------------------------
// Tiled WMMA GEMM: Y[m][n] = sum_k X[m][k] * W[n][k] + bias[n]
// M = 4096, N = 1024, K = 1024. Block tile 128x128, 8 waves (2M x 4N),
// each wave computes 64x32 via 4x2 grid of 16x16x32 WMMAs.
// Double-buffered LDS fed by async-to-LDS copies (one-iteration lookahead).
// mode 0: dst f16 head-major [b,h,s,d]
// mode 1: dst f16 transposed  [b,h,d,s]
// mode 2: dst f32 row-major   [m][n]     (final output projection)
// ---------------------------------------------------------------------------
#define LSTR 40  // LDS row stride in halves (80B, 16B aligned, pads banks)
#define KCHUNKS (HID / 32)

__global__ __launch_bounds__(256) void gemm_wmma_f16(
    const _Float16* __restrict__ X, const _Float16* __restrict__ W,
    const float* __restrict__ bias, void* __restrict__ dst, int mode) {
  __shared__ _Float16 Xl[2][128 * LSTR];
  __shared__ _Float16 Wl[2][128 * LSTR];

  const int tid  = threadIdx.x;
  const int wv   = tid >> 5;
  const int lane = tid & 31;
  const int hs   = lane >> 4;   // lane half select
  const int r    = lane & 15;
  const int wm   = wv & 1;      // 0..1  (64-row slots)
  const int wn   = wv >> 1;     // 0..3  (32-col slots)
  const int mbase = blockIdx.y * 128;
  const int nbase = blockIdx.x * 128;

  v8f acc[4][2];
#pragma unroll
  for (int mi = 0; mi < 4; ++mi)
#pragma unroll
    for (int ni = 0; ni < 2; ++ni) acc[mi][ni] = (v8f){};

  const int srow = tid >> 1;        // 0..127
  const int seg  = (tid & 1) * 16;  // 0 or 16 halves

  const _Float16* gx0 = X + (size_t)(mbase + srow) * HID + seg;
  const _Float16* gw0 = W + (size_t)(nbase + srow) * HID + seg;

  // Prologue: issue chunk 0 into buffer 0 (4 async b128 per thread).
  {
    _Float16* lx = &Xl[0][srow * LSTR + seg];
    _Float16* lw = &Wl[0][srow * LSTR + seg];
    async_ld_b128(lx,     gx0);
    async_ld_b128(lx + 8, gx0 + 8);
    async_ld_b128(lw,     gw0);
    async_ld_b128(lw + 8, gw0 + 8);
  }

  for (int it = 0; it < KCHUNKS; ++it) {
    const int bsel = it & 1;
    if (it + 1 < KCHUNKS) {
      // Issue next chunk into the other buffer, then wait for this chunk's
      // 4 ops (async ops complete in order: <=4 outstanding leaves only the
      // lookahead chunk in flight).
      const int nb = (it + 1) & 1;
      const int kn = (it + 1) * 32;
      _Float16* lx = &Xl[nb][srow * LSTR + seg];
      _Float16* lw = &Wl[nb][srow * LSTR + seg];
      async_ld_b128(lx,     gx0 + kn);
      async_ld_b128(lx + 8, gx0 + kn + 8);
      async_ld_b128(lw,     gw0 + kn);
      async_ld_b128(lw + 8, gw0 + kn + 8);
      wait_async_le4();
    } else {
      wait_async_0();
    }
    __syncthreads();

    v16h bfr[2];
#pragma unroll
    for (int ni = 0; ni < 2; ++ni)
      bfr[ni] =
          frag_from_halves(&Wl[bsel][(wn * 32 + ni * 16 + r) * LSTR + hs * 8]);

#pragma unroll
    for (int mi = 0; mi < 4; ++mi) {
      v16h afr =
          frag_from_halves(&Xl[bsel][(wm * 64 + mi * 16 + r) * LSTR + hs * 8]);
#pragma unroll
      for (int ni = 0; ni < 2; ++ni)
        acc[mi][ni] = wmma_f16(afr, bfr[ni], acc[mi][ni]);
    }
    __syncthreads();  // everyone done reading bsel before it is overwritten
  }

  // Epilogue: C/D layout -> element (M = i + 8*hs, N = lane&15) per VGPR i.
#pragma unroll
  for (int mi = 0; mi < 4; ++mi) {
#pragma unroll
    for (int ni = 0; ni < 2; ++ni) {
      const int C  = nbase + wn * 32 + ni * 16 + r;
      const float bv = bias[C];
      const int R0 = mbase + wm * 64 + mi * 16 + hs * 8;
#pragma unroll
      for (int i = 0; i < 8; ++i) {
        const int R = R0 + i;
        const float val = acc[mi][ni][i] + bv;
        if (mode == 2) {
          ((float*)dst)[(size_t)R * HID + C] = val;
        } else {
          const int b = R >> 11, s = R & (SEQ - 1);
          const int h = C >> 6, d = C & (DK - 1);
          size_t idx;
          if (mode == 0)
            idx = (((size_t)(b * NHEAD + h) * SEQ + s) * DK + d);
          else  // mode 1: V transposed [b,h,d,s]
            idx = (((size_t)(b * NHEAD + h) * DK + d) * SEQ + s);
          ((_Float16*)dst)[idx] = (_Float16)val;
        }
      }
    }
  }
}

// ---------------------------------------------------------------------------
// Attention: one workgroup per (b*NH, 16-row q strip).
// Phase 1: scores (16 x 2048) via WMMA, scale + mask, to LDS (f32).
// Phase 2: row softmax; normalized probs -> LDS and -> attn output (f32).
// Phase 3: O = P @ V via WMMA, split-K x2 with LDS reduce; ctx f16 out.
// LDS: 128KB scores + ~5KB scratch = ~133KB (< 320KB WGP LDS).
// ---------------------------------------------------------------------------
__global__ __launch_bounds__(256) void attention_wmma(
    const _Float16* __restrict__ qh, const _Float16* __restrict__ kh,
    const _Float16* __restrict__ vT, const int* __restrict__ mask,
    float* __restrict__ attn_out, _Float16* __restrict__ ctx) {
  __shared__ float sc[16 * SEQ];     // 131072 B
  __shared__ float red[16 * 16];
  __shared__ float mrow[16];
  __shared__ float lrow[16];
  __shared__ float part[4 * 256];    // split-K partials (waves 4..7)

  const int bh = blockIdx.y;            // 0..31
  const int b  = bh >> 4;
  const int h  = bh & (NHEAD - 1);
  const int qb = blockIdx.x * 16;       // q row base
  const int tid  = threadIdx.x;
  const int wv   = tid >> 5;
  const int lane = tid & 31;
  const int hs   = lane >> 4;
  const int r    = lane & 15;
  const float NEG = -__builtin_huge_valf();

  // ---- Q fragments (reused by all 16 column tiles of this wave) ----
  const _Float16* qrow = qh + ((size_t)bh * SEQ + qb + r) * DK;
  const v16h a0 = frag_from_halves(qrow + hs * 8);        // d 0..31
  const v16h a1 = frag_from_halves(qrow + 32 + hs * 8);   // d 32..63

  // ---- Phase 1: scores. Wave wv covers KV columns [wv*256, wv*256+256) ----
  for (int jj = 0; jj < 16; ++jj) {
    const int cb = wv * 256 + jj * 16;
    const _Float16* krow = kh + ((size_t)bh * SEQ + cb + r) * DK;
    const v16h b0 = frag_from_halves(krow + hs * 8);
    const v16h b1 = frag_from_halves(krow + 32 + hs * 8);
    v8f acc = (v8f){};
    acc = wmma_f16(a0, b0, acc);
    acc = wmma_f16(a1, b1, acc);
    const int mv = mask[b * SEQ + cb + r];   // per-lane column mask
#pragma unroll
    for (int i = 0; i < 8; ++i) {
      float s = acc[i] * 0.125f;             // 1/sqrt(DK)
      if (mv == 0) s = NEG;
      sc[(i + hs * 8) * SEQ + cb + r] = s;
    }
  }
  __syncthreads();

  // ---- Phase 2: softmax over each of the 16 rows (16 threads per row) ----
  const int row = tid >> 4;       // 0..15
  const int sub = tid & 15;       // 0..15
  float mx = NEG;
  for (int j = 0; j < 128; ++j) mx = fmaxf(mx, sc[row * SEQ + sub + j * 16]);
  red[row * 16 + sub] = mx;
  __syncthreads();
  if (sub == 0) {
    float m = red[row * 16];
    for (int t = 1; t < 16; ++t) m = fmaxf(m, red[row * 16 + t]);
    mrow[row] = m;
  }
  __syncthreads();
  const float m = mrow[row];
  float sum = 0.f;
  for (int j = 0; j < 128; ++j)
    sum += __expf(sc[row * SEQ + sub + j * 16] - m);
  red[row * 16 + sub] = sum;
  __syncthreads();
  if (sub == 0) {
    float l = 0.f;
    for (int t = 0; t < 16; ++t) l += red[row * 16 + t];
    lrow[row] = l;
  }
  __syncthreads();
  const float inv = 1.0f / lrow[row];
  float* aout = attn_out + ((size_t)bh * SEQ + qb + row) * SEQ;
  for (int j = 0; j < 128; ++j) {
    const int c = sub + j * 16;
    const float p = __expf(sc[row * SEQ + c] - m) * inv;
    sc[row * SEQ + c] = p;      // keep normalized prob for PV
    aout[c] = p;                // the one mandatory HBM write of attn
  }
  __syncthreads();

  // ---- Phase 3: O = P @ V. 8 waves = 4 d-tiles x 2 K-halves ----
  const int nt = wv & 3;         // d tile (16 cols of DK)
  const int kh2 = wv >> 2;       // K half
  v8f o = (v8f){};
  for (int kc = 0; kc < 32; ++kc) {
    const int kb = kh2 * 1024 + kc * 32;
    // A fragment from f32 probs in LDS (convert to f16)
    const float* prow = &sc[r * SEQ + kb + hs * 8];
    v16h pa;
#pragma unroll
    for (int i = 0; i < 8; ++i) pa[i] = (_Float16)prow[i];
#pragma unroll
    for (int i = 0; i < 8; ++i) pa[8 + i] = (_Float16)prow[16 + i];
    // B fragment from transposed V: row d, contiguous in k
    const _Float16* vrow = vT + ((size_t)bh * DK + nt * 16 + r) * SEQ + kb;
    const v16h vb = frag_from_halves(vrow + hs * 8);
    o = wmma_f16(pa, vb, o);
  }
  if (wv >= 4) {
#pragma unroll
    for (int i = 0; i < 8; ++i)
      part[nt * 256 + (i + hs * 8) * 16 + r] = o[i];
  }
  __syncthreads();
  if (wv < 4) {
#pragma unroll
    for (int i = 0; i < 8; ++i) {
      const float val = o[i] + part[nt * 256 + (i + hs * 8) * 16 + r];
      const int q = qb + i + hs * 8;
      ctx[((size_t)(b * SEQ + q)) * HID + h * DK + nt * 16 + r] = (_Float16)val;
    }
  }
}

// ---------------------------------------------------------------------------
// Host launcher
// ---------------------------------------------------------------------------
extern "C" void kernel_launch(void* const* d_in, const int* in_sizes, int n_in,
                              void* d_out, int out_size, void* d_ws,
                              size_t ws_size, hipStream_t stream) {
  (void)in_sizes; (void)n_in; (void)out_size; (void)ws_size;

  const float* q   = (const float*)d_in[0];
  const float* k   = (const float*)d_in[1];
  const float* v   = (const float*)d_in[2];
  const int*   msk = (const int*)  d_in[3];
  const float* Wq  = (const float*)d_in[4];
  const float* bq  = (const float*)d_in[5];
  const float* Wk  = (const float*)d_in[6];
  const float* bk  = (const float*)d_in[7];
  const float* Wv  = (const float*)d_in[8];
  const float* bv  = (const float*)d_in[9];
  const float* Wo  = (const float*)d_in[10];
  const float* bo  = (const float*)d_in[11];

  float* out  = (float*)d_out;                              // [B,S,H]
  float* attn = out + (size_t)BATCH * SEQ * HID;            // [B,NH,S,S]

  const size_t NE = (size_t)BATCH * SEQ * HID;   // 4,194,304
  const size_t WE = (size_t)HID * HID;           // 1,048,576

  char* base = (char*)d_ws;
  size_t off = 0;
  auto take = [&](size_t bytes) -> void* {
    void* p = base + off;
    off += (bytes + 255) & ~(size_t)255;
    return p;
  };
  _Float16* q16  = (_Float16*)take(NE * 2);
  _Float16* k16  = (_Float16*)take(NE * 2);
  _Float16* v16  = (_Float16*)take(NE * 2);
  _Float16* wq16 = (_Float16*)take(WE * 2);
  _Float16* wk16 = (_Float16*)take(WE * 2);
  _Float16* wv16 = (_Float16*)take(WE * 2);
  _Float16* wo16 = (_Float16*)take(WE * 2);
  _Float16* qhp  = (_Float16*)take(NE * 2);   // [b,h,s,d]
  _Float16* khp  = (_Float16*)take(NE * 2);   // [b,h,s,d]
  _Float16* vTp  = (_Float16*)take(NE * 2);   // [b,h,d,s]
  _Float16* ctx  = (_Float16*)take(NE * 2);   // [b,s,H]

  // 1) converts
  const int cvtBlk = 256;
  hipLaunchKernelGGL(cvt_f32_to_f16, dim3(4096), dim3(cvtBlk), 0, stream, q, q16, (int)NE);
  hipLaunchKernelGGL(cvt_f32_to_f16, dim3(4096), dim3(cvtBlk), 0, stream, k, k16, (int)NE);
  hipLaunchKernelGGL(cvt_f32_to_f16, dim3(4096), dim3(cvtBlk), 0, stream, v, v16, (int)NE);
  hipLaunchKernelGGL(cvt_f32_to_f16, dim3(1024), dim3(cvtBlk), 0, stream, Wq, wq16, (int)WE);
  hipLaunchKernelGGL(cvt_f32_to_f16, dim3(1024), dim3(cvtBlk), 0, stream, Wk, wk16, (int)WE);
  hipLaunchKernelGGL(cvt_f32_to_f16, dim3(1024), dim3(cvtBlk), 0, stream, Wv, wv16, (int)WE);
  hipLaunchKernelGGL(cvt_f32_to_f16, dim3(1024), dim3(cvtBlk), 0, stream, Wo, wo16, (int)WE);

  // 2) projections (M=4096 rows, N=1024, K=1024)
  dim3 ggrid(HID / 128, (BATCH * SEQ) / 128);  // (8, 32)
  hipLaunchKernelGGL(gemm_wmma_f16, ggrid, dim3(256), 0, stream, q16, wq16, bq, (void*)qhp, 0);
  hipLaunchKernelGGL(gemm_wmma_f16, ggrid, dim3(256), 0, stream, k16, wk16, bk, (void*)khp, 0);
  hipLaunchKernelGGL(gemm_wmma_f16, ggrid, dim3(256), 0, stream, v16, wv16, bv, (void*)vTp, 1);

  // 3) attention
  dim3 agrid(SEQ / 16, BATCH * NHEAD);         // (128, 32)
  hipLaunchKernelGGL(attention_wmma, agrid, dim3(256), 0, stream,
                     qhp, khp, vTp, msk, attn, ctx);

  // 4) output projection -> f32 d_out
  hipLaunchKernelGGL(gemm_wmma_f16, ggrid, dim3(256), 0, stream, ctx, wo16, bo, d_out, 2);
}
